// BalancedPrototypeLoss_78219944394820
// MI455X (gfx1250) — matrix operand order repaired and checked by reading, not compiled, not run on hardware
//
#include <hip/hip_runtime.h>
#include <hip/hip_bf16.h>
#include <math.h>

typedef __attribute__((ext_vector_type(16))) __bf16 v16bf;
typedef __attribute__((ext_vector_type(8)))  __bf16 v8bf;
typedef __attribute__((ext_vector_type(8)))  float  v8f;

#define MARGIN_C        0.3f
#define CLST_SCALE_C    0.8f
#define SEP_SCALE_C     0.08f
#define DIV_SCALE_C     0.01f
#define CONTRAST_SCALE_C 0.1f

// acc layout (floats): [0]=contrastive_sum, [1..C]=cls_n, [1+C..]=cls_sum,
// [1+2C..]=sep_sum, [1+3C..]=cls_pair_sum

// ---------------------------------------------------------------------------
// Kernel 1: streaming pass over similarities (memory-bound, 65.5 MB)
// One block per sample. Coalesced load of the 1000-float row into LDS,
// per-class min over P, tree-reduce for other-class min, 3 atomics.
// ---------------------------------------------------------------------------
__global__ void sample_min_kernel(const float* __restrict__ sim,
                                  const int* __restrict__ labels,
                                  const int* __restrict__ pidx,
                                  float* __restrict__ acc,
                                  int C, int P) {
    extern __shared__ float smem[];
    float* data = smem;            // C*P
    float* minv = data + C * P;    // C
    float* red  = minv + C;        // blockDim.x
    const int b   = blockIdx.x;
    const int tid = threadIdx.x;
    const float* row = sim + (size_t)b * C * P;

    for (int i = tid; i < C * P; i += blockDim.x) data[i] = row[i];
    __syncthreads();

    for (int c = tid; c < C; c += blockDim.x) {
        int cnt = pidx[2 * c + 1] - pidx[2 * c];
        int lim = cnt < P ? cnt : P;
        float m = __builtin_inff();
        for (int p = 0; p < lim; ++p) m = fminf(m, 1.0f - data[c * P + p]);
        minv[c] = m;
    }
    __syncthreads();

    const int label = labels[b];
    float x = __builtin_inff();
    for (int c = tid; c < C; c += blockDim.x)
        if (c != label) x = fminf(x, minv[c]);
    red[tid] = x;
    __syncthreads();
    for (int s = blockDim.x >> 1; s > 0; s >>= 1) {
        if (tid < s) red[tid] = fminf(red[tid], red[tid + s]);
        __syncthreads();
    }
    if (tid == 0) {
        float own = minv[label];
        float sep = fmaxf(MARGIN_C - red[0], 0.0f);
        atomicAdd(&acc[1 + label], 1.0f);
        atomicAdd(&acc[1 + C + label], own);
        atomicAdd(&acc[1 + 2 * C + label], sep);
    }
}

// ---------------------------------------------------------------------------
// Kernel 2: row-normalize prototypes, emit bf16 (padded rows zeroed)
// ---------------------------------------------------------------------------
__global__ void normalize_kernel(const float* __restrict__ protos,
                                 __bf16* __restrict__ pn,
                                 int T, int D) {
    __shared__ float red[128];
    const int r = blockIdx.x;
    const int tid = threadIdx.x;
    if (r >= T) {
        for (int k = tid; k < D; k += blockDim.x) pn[(size_t)r * D + k] = (__bf16)0.0f;
        return;
    }
    const float* src = protos + (size_t)r * D;
    float s = 0.0f;
    for (int k = tid; k < D; k += blockDim.x) { float v = src[k]; s += v * v; }
    red[tid] = s;
    __syncthreads();
    for (int st = 64; st > 0; st >>= 1) {
        if (tid < st) red[tid] += red[tid + st];
        __syncthreads();
    }
    const float inv = 1.0f / fmaxf(sqrtf(red[0]), 1e-12f);
    for (int k = tid; k < D; k += blockDim.x)
        pn[(size_t)r * D + k] = (__bf16)(src[k] * inv);
}

// ---------------------------------------------------------------------------
// Kernel 3: proto_class[t] = searchsorted(starts, t, 'right') - 1
// ---------------------------------------------------------------------------
__global__ void pclass_kernel(const int* __restrict__ pidx,
                              int* __restrict__ pclass, int C, int T) {
    const int t = blockIdx.x * blockDim.x + threadIdx.x;
    if (t >= T) return;
    int cls = -1;
    for (int c = 0; c < C; ++c) cls += (pidx[2 * c] <= t) ? 1 : 0;
    pclass[t] = cls;
}

// ---------------------------------------------------------------------------
// Kernel 4: sim = pn @ pn^T via v_wmma_f32_16x16x32_bf16, fused loss epilogue.
// One wave per 16x16 output tile. EXEC is all-ones through the WMMA loop.
// A fragment (16x32 bf16): lane m=lane&15 is row i16+m; lanes<16 hold
// K={0..7,16..23}, lanes>=16 hold K={8..15,24..31} (two 16B contiguous loads).
// B fragment (32x16 bf16): B = pn^T, so lane n=lane&15 loads row j16+n with
// the same K order as A (lanes<16: K {0..7,16..23}, lanes>=16: K {8..15,24..31}).
// ---------------------------------------------------------------------------
__global__ void proto_sim_kernel(const __bf16* __restrict__ pn,
                                 const int* __restrict__ pclass,
                                 const unsigned char* __restrict__ vm,
                                 float* __restrict__ acc,
                                 int T, int D, int C) {
    const int i16  = blockIdx.y * 16;
    const int j16  = blockIdx.x * 16;
    const int lane = threadIdx.x;   // 0..31 (wave32)
    const int half = lane >> 4;     // 0 or 1
    const int m    = lane & 15;

    const __bf16* arow = pn + (size_t)(i16 + m) * D;
    const __bf16* brow = pn + (size_t)(j16 + m) * D;

    v8f cacc = {};
    for (int k0 = 0; k0 < D; k0 += 32) {
        v8bf alo = *(const v8bf*)(arow + k0 + half * 8);
        v8bf ahi = *(const v8bf*)(arow + k0 + 16 + half * 8);
        v16bf a;
#pragma unroll
        for (int q = 0; q < 8; ++q) { a[q] = alo[q]; a[8 + q] = ahi[q]; }
        v8bf blo = *(const v8bf*)(brow + k0 + half * 8);
        v8bf bhi = *(const v8bf*)(brow + k0 + 16 + half * 8);
        v16bf b;
#pragma unroll
        for (int q = 0; q < 8; ++q) { b[q] = blo[q]; b[8 + q] = bhi[q]; }
        cacc = __builtin_amdgcn_wmma_f32_16x16x32_bf16(
            false, a, false, b, (short)0, cacc, false, false);
    }

    // Fused epilogue: C/D layout — VGPR v: lanes<16 -> row i16+v, lanes>=16 ->
    // row i16+v+8; column j16+(lane&15).
    const int cc   = j16 + m;
    const bool ccin = cc < T;
    const int  ccls = ccin ? pclass[cc] : -2;
    const bool ccvm = ccin ? (vm[cc] != 0) : false;
    float csum = 0.0f;
#pragma unroll
    for (int v = 0; v < 8; ++v) {
        const int r = i16 + v + half * 8;
        const float sv = cacc[v];
        if (r < T && ccin && r != cc) {
            if (ccvm && vm[r]) csum += sv;                       // contrastive
            if (ccls >= 0 && pclass[r] == ccls) {                // same-class pair
                float rel = sv - 0.5f;
                if (rel > 0.0f) atomicAdd(&acc[1 + 3 * C + ccls], rel);
            }
        }
    }
    // wave32 reduce, single atomic per wave
    for (int off = 16; off > 0; off >>= 1) csum += __shfl_down(csum, off, 32);
    if (lane == 0) atomicAdd(&acc[0], csum);
}

// ---------------------------------------------------------------------------
// Kernel 5: finalize the 5 outputs (single block)
// ---------------------------------------------------------------------------
__global__ void finalize_kernel(const float* __restrict__ acc,
                                const int* __restrict__ pidx,
                                const unsigned char* __restrict__ vm,
                                float* __restrict__ out,
                                int C, int T) {
    __shared__ float red[128];
    const int tid = threadIdx.x;
    float clusterP = 0.f, sepP = 0.f, divP = 0.f, hasF = 0.f, dvF = 0.f;
    if (tid < C) {
        float n = acc[1 + tid];
        float nd = fmaxf(n, 1.0f);
        if (n > 0.0f) {
            clusterP = (1.0f / sqrtf(n + 1e-6f)) * (acc[1 + C + tid] / nd);
            sepP     = acc[1 + 2 * C + tid] / nd;
            hasF     = 1.0f;
        }
        float cnt = (float)(pidx[2 * tid + 1] - pidx[2 * tid]);
        float np  = cnt * (cnt - 1.0f);
        if (cnt > 1.0f) { divP = acc[1 + 3 * C + tid] / fmaxf(np, 1.0f); dvF = 1.0f; }
    }
    float sums[5] = {clusterP, sepP, divP, hasF, dvF};
    float tot[5];
    for (int q = 0; q < 5; ++q) {
        red[tid] = sums[q];
        __syncthreads();
        for (int s = 64; s > 0; s >>= 1) {
            if (tid < s) red[tid] += red[tid + s];
            __syncthreads();
        }
        tot[q] = red[0];
        __syncthreads();
    }
    float s = 0.0f;
    for (int t = tid; t < T; t += blockDim.x) s += vm[t] ? 1.0f : 0.0f;
    red[tid] = s;
    __syncthreads();
    for (int st = 64; st > 0; st >>= 1) {
        if (tid < st) red[tid] += red[tid + st];
        __syncthreads();
    }
    if (tid == 0) {
        float S       = red[0];
        float nvalid  = fmaxf(tot[3], 1.0f);
        float ndv     = fmaxf(tot[4], 1.0f);
        float cluster = tot[0] / nvalid * CLST_SCALE_C;
        float sep     = tot[1] / nvalid * SEP_SCALE_C;
        float divl    = tot[2] / ndv * DIV_SCALE_C;
        float nvp     = fmaxf(S * S - S, 1.0f);
        float contr   = acc[0] / nvp * CONTRAST_SCALE_C;
        out[0] = cluster;
        out[1] = sep;
        out[2] = divl;
        out[3] = contr;
        out[4] = cluster + sep + divl + contr;
    }
}

// ---------------------------------------------------------------------------
extern "C" void kernel_launch(void* const* d_in, const int* in_sizes, int n_in,
                              void* d_out, int out_size, void* d_ws, size_t ws_size,
                              hipStream_t stream) {
    const float* sim            = (const float*)d_in[0];
    const int* labels           = (const int*)d_in[1];
    const float* protos         = (const float*)d_in[2];
    const int* pidx             = (const int*)d_in[3];
    const unsigned char* vmask  = (const unsigned char*)d_in[4];
    float* out                  = (float*)d_out;

    const int B = in_sizes[1];
    const int T = in_sizes[4];
    const int D = in_sizes[2] / T;
    const int C = in_sizes[3] / 2;
    const int P = in_sizes[0] / (B * C);
    const int Tpad = (T + 15) & ~15;

    // workspace layout
    float* acc = (float*)d_ws;                                  // 1 + 4*C floats
    size_t accBytes = (size_t)(1 + 4 * C) * sizeof(float);
    size_t off = (accBytes + 255) & ~(size_t)255;
    int* pclass = (int*)((char*)d_ws + off);
    off = (off + (size_t)Tpad * sizeof(int) + 255) & ~(size_t)255;
    __bf16* pn = (__bf16*)((char*)d_ws + off);                  // Tpad * D bf16

    hipMemsetAsync(acc, 0, accBytes, stream);

    const int threads1 = 128;
    const size_t smem1 = (size_t)(C * P + C + threads1) * sizeof(float);
    sample_min_kernel<<<B, threads1, smem1, stream>>>(sim, labels, pidx, acc, C, P);

    normalize_kernel<<<Tpad, 128, 0, stream>>>(protos, pn, T, D);
    pclass_kernel<<<(T + 127) / 128, 128, 0, stream>>>(pidx, pclass, C, T);

    dim3 grid(Tpad / 16, Tpad / 16);
    proto_sim_kernel<<<grid, 32, 0, stream>>>(pn, pclass, vmask, acc, T, D, C);

    finalize_kernel<<<1, 128, 0, stream>>>(acc, pidx, vmask, out, C, T);
}
